// Dep_Context_5420248727989
// MI455X (gfx1250) — matrix operand order repaired and checked by reading, not compiled, not run on hardware
//
#include <hip/hip_runtime.h>
#include <hip/hip_bf16.h>

// ---------------------------------------------------------------------------
// CDNA5 (gfx1250, wave32) fused attention pipeline.
// - flash-style softmax-attention on v_wmma_f32_16x16x32_f16 with
//   double-buffered async global->LDS staging (global_load_async_to_lds_b128,
//   s_wait_asynccnt), 32 queries per wave.
// - value conv1x1 (Wv @ p, the traffic-heavy GEMM) also on WMMA, with the
//   64KB A-operand staged per-WG via the Tensor Data Mover
//   (tensor_load_to_lds + s_wait_tensorcnt) when available.
// ---------------------------------------------------------------------------

typedef __attribute__((ext_vector_type(16))) _Float16 v16h;
typedef __attribute__((ext_vector_type(8)))  float    v8f;
typedef __attribute__((ext_vector_type(8)))  int      v8i;
typedef __attribute__((ext_vector_type(4)))  int      v4i;
typedef __attribute__((ext_vector_type(4)))  unsigned v4u;

static __device__ __forceinline__ v8f wmma_f16(v16h a, v16h b, v8f c) {
  return __builtin_amdgcn_wmma_f32_16x16x32_f16(false, a, false, b,
                                                (short)0, c, false, false);
}

static __device__ __forceinline__ int pack2h(float lo, float hi) {
  _Float16 a = (_Float16)lo, b = (_Float16)hi;
  unsigned short ua = __builtin_bit_cast(unsigned short, a);
  unsigned short ub = __builtin_bit_cast(unsigned short, b);
  return (int)((unsigned)ua | ((unsigned)ub << 16));
}

// LDS byte offset (addrspace(3) is 32-bit, 0-based from LDS_BASE).
typedef __attribute__((address_space(3))) int lds_int;
static __device__ __forceinline__ unsigned lds_off(const void* p) {
  return (unsigned)(unsigned long long)(lds_int*)p;
}

static __device__ __forceinline__ void async_ld_b128(unsigned lds_byte, unsigned long long gaddr) {
  asm volatile("global_load_async_to_lds_b128 %0, %1, off"
               :: "v"(lds_byte), "v"(gaddr) : "memory");
}
static __device__ __forceinline__ void wait_async0() {
  asm volatile("s_wait_asynccnt 0x0" ::: "memory");
}

#define L 4096
#define HP 64

// ---------------------------------------------------------------------------
// 1. AvgPool2d(3, stride 2, VALID): [NC,129,129] -> [NC,64,64]
// ---------------------------------------------------------------------------
__global__ void pool_kernel(const float* __restrict__ in, float* __restrict__ out, int NC) {
  int idx = blockIdx.x * 256 + threadIdx.x;
  if (idx >= NC * HP * HP) return;
  int x = idx & 63, y = (idx >> 6) & 63, c = idx >> 12;
  const float* src = in + (size_t)c * (129 * 129) + (2 * y) * 129 + 2 * x;
  float s = 0.f;
#pragma unroll
  for (int dy = 0; dy < 3; ++dy)
#pragma unroll
    for (int dx = 0; dx < 3; ++dx) s += src[dy * 129 + dx];
  out[idx] = s * (1.f / 9.f);
}

// ---------------------------------------------------------------------------
// 2. Query/Key 1x1 convs -> WMMA operand layouts (f16, K padded 18->32).
// ---------------------------------------------------------------------------
__global__ void qkprep_kernel(const float* __restrict__ u, const float* __restrict__ p,
                              const float* __restrict__ Wq, const float* __restrict__ Wk,
                              int* __restrict__ Qop, int* __restrict__ Kop) {
  int idx = blockIdx.x * 256 + threadIdx.x;
  if (idx >= 4 * L) return;
  int l = idx & (L - 1), n = idx >> 12;
  int x = l & 63, y = l >> 6;

  float coord[8];
  coord[0] = x * (2.f / 64.f) - 1.f;
  coord[1] = y * (2.f / 64.f) - 1.f;
  coord[2] = (x + 1) * (2.f / 64.f) - 1.f;
  coord[3] = (y + 1) * (2.f / 64.f) - 1.f;
  coord[4] = 0.5f * (coord[0] + coord[2]);
  coord[5] = 0.5f * (coord[1] + coord[3]);
  coord[6] = 1.f / 64.f;
  coord[7] = 1.f / 64.f;

  float qin[18];
#pragma unroll
  for (int c = 0; c < 10; ++c) qin[c] = u[((size_t)(n * 10 + c) << 12) + l];
#pragma unroll
  for (int c = 0; c < 8; ++c) qin[10 + c] = coord[c];
  float qv[18];
#pragma unroll
  for (int o = 0; o < 18; ++o) {
    float s = 0.f;
#pragma unroll
    for (int i = 0; i < 18; ++i) s += Wq[o * 18 + i] * qin[i];
    qv[o] = s;
  }

  float kv[18];
#pragma unroll
  for (int o = 0; o < 18; ++o) kv[o] = 0.f;
  for (int i = 0; i < 256; ++i) {
    float pv = p[((size_t)(n * 256 + i) << 12) + l];
#pragma unroll
    for (int o = 0; o < 18; ++o) kv[o] += Wk[o * 264 + i] * pv;
  }
#pragma unroll
  for (int c = 0; c < 8; ++c) {
    float pv = coord[c];
#pragma unroll
    for (int o = 0; o < 18; ++o) kv[o] += Wk[o * 264 + 256 + c] * pv;
  }

  int chunk = l >> 4, j = l & 15;
  int* qlo = Qop + (((size_t)(n * 256 + chunk) * 32 + j) * 8);
  int* qhi = qlo + 16 * 8;
#pragma unroll
  for (int v = 0; v < 8; ++v) qlo[v] = pack2h(qv[2 * v], qv[2 * v + 1]);
  qhi[0] = pack2h(qv[16], qv[17]);
#pragma unroll
  for (int v = 1; v < 8; ++v) qhi[v] = 0;

  int* klo = Kop + (((size_t)(n * 256 + chunk) * 32 + j) * 8);
  int* khi = klo + 16 * 8;
#pragma unroll
  for (int v = 0; v < 8; ++v) klo[v] = pack2h(kv[2 * v], kv[2 * v + 1]);
  khi[0] = pack2h(kv[16], kv[17]);
#pragma unroll
  for (int v = 1; v < 8; ++v) khi[v] = 0;
}

// ---------------------------------------------------------------------------
// 3a. Pack Wv[128,256] -> A-operand tiles Wop[kt=8][mt=8][lane=32][v=8] dwords.
//     A 16x32 f16: lanes<16: v = K (2v,2v+1);
//                  lanes>=16: v = K (16+2v,17+2v).
// ---------------------------------------------------------------------------
__global__ void wpack_kernel(const float* __restrict__ Wv, int* __restrict__ Wop) {
  int idx = blockIdx.x * 256 + threadIdx.x;  // 16384
  int v = idx & 7, lane = (idx >> 3) & 31, mt = (idx >> 8) & 7, kt = (idx >> 11) & 7;
  int hi = lane >> 4;
  int kbase = hi * 16 + 2 * v;
  int c = mt * 16 + (lane & 15);
  int ic = kt * 32 + kbase;
  Wop[idx] = pack2h(Wv[c * 256 + ic], Wv[c * 256 + ic + 1]);
}

// ---------------------------------------------------------------------------
// 3b. Pack p -> B-operand tiles Pop[n][kt=8][lt=256][lane=32][v=8] dwords.
//     B 32x16 f16: lanes<16: v = K (2v,2v+1); lanes>=16: v = K (16+2v,17+2v);
//     lane%16 = column (pixel).
// ---------------------------------------------------------------------------
__global__ void ppack_kernel(const float* __restrict__ p, int* __restrict__ Pop) {
  int idx = blockIdx.x * 256 + threadIdx.x;  // 4*8*256*32*8 = 2M
  int v = idx & 7, lane = (idx >> 3) & 31, lt = (idx >> 8) & 255;
  int kt = (idx >> 16) & 7, n = idx >> 19;
  int l = lt * 16 + (lane & 15);
  int ic = kt * 32 + ((lane >= 16) ? 16 : 0) + 2 * v;
  const float* pp = p + ((size_t)(n * 256 + ic) << 12) + l;
  Pop[idx] = pack2h(pp[0], pp[1 << 12]);
}

// ---------------------------------------------------------------------------
// 3c. vraw[n][128][L] = Wv @ p on WMMA. Each wave: one 16-pixel strip x all
//     128 channels (64 wmmas). Whole 64KB A-operand staged in LDS per WG,
//     via TDM (tensor_load_to_lds) when the builtin exists, else async b128.
// ---------------------------------------------------------------------------
__global__ __launch_bounds__(256) void vgemm_kernel(const int* __restrict__ Wop,
                                                    const int* __restrict__ Pop,
                                                    float* __restrict__ vraw) {
  __shared__ int ldsw[16384];  // 64KB: [kt*8+mt][lane][v]
  const int lane = threadIdx.x & 31;
  const int wave = threadIdx.x >> 5;
  const int nb = blockIdx.x >> 5;                  // batch
  const int ltile = (blockIdx.x & 31) * 8 + wave;  // 16-pixel strip

  const unsigned ldsbase = lds_off(&ldsw[0]);
  const unsigned long long ga = (unsigned long long)Wop;

#if __has_builtin(__builtin_amdgcn_tensor_load_to_lds)
  if (wave == 0) {
    // D# group0: count=1, lds_addr, global_addr (57b), type=2 ("image")
    v4u g0;
    g0[0] = 1u;
    g0[1] = ldsbase;
    g0[2] = (unsigned)ga;
    g0[3] = ((unsigned)(ga >> 32) & 0x01FFFFFFu) | 0x80000000u;
    // D# group1: data_size=4B, tensor_dim0=16384, tensor_dim1=1,
    // tile_dim0=16384 (1-D tile), tile_dim1=0 (unused), stride0=16384.
    v8i g1;
    g1[0] = 0x20000;                 // data_size=2 (4B)
    g1[1] = (int)(16384u << 16);     // tensor_dim0[15:0] in [31:16]
    g1[2] = 1 << 16;                 // tensor_dim0 hi=0, tensor_dim1 lo=1
    g1[3] = (int)(16384u << 16);     // tensor_dim1 hi=0, tile_dim0=16384
    g1[4] = 0;                       // tile_dim1=0, tile_dim2=0
    g1[5] = 16384;                   // tensor_dim0_stride lo
    g1[6] = 0;
    g1[7] = 0;
    v4i gz = {0, 0, 0, 0};
#if defined(__clang_major__) && __clang_major__ >= 23
    v8i gz8 = {0, 0, 0, 0, 0, 0, 0, 0};
    __builtin_amdgcn_tensor_load_to_lds(g0, g1, gz, gz, gz8, 0);
#else
    __builtin_amdgcn_tensor_load_to_lds(g0, g1, gz, gz, 0);
#endif
    __builtin_amdgcn_s_wait_tensorcnt(0);
  }
  __syncthreads();
#else
  {
#pragma unroll
    for (int i = 0; i < 16; ++i) {
      unsigned t = (unsigned)((i * 256 + (int)threadIdx.x) * 16);
      async_ld_b128(ldsbase + t, ga + t);
    }
    wait_async0();
    __syncthreads();
  }
#endif

  v8f o[8];
#pragma unroll
  for (int m = 0; m < 8; ++m) o[m] = (v8f){};

#pragma unroll 1
  for (int kt = 0; kt < 8; ++kt) {
    v8i vb;
    const int* bp = Pop + (((size_t)(nb * 8 + kt) * 256 + ltile) * 32 + lane) * 8;
#pragma unroll
    for (int v = 0; v < 8; ++v) vb[v] = bp[v];
    v16h bh = __builtin_bit_cast(v16h, vb);
    const int* al = &ldsw[(kt * 8) * 256 + lane * 8];
#pragma unroll
    for (int mt = 0; mt < 8; ++mt) {
      v8i a;
#pragma unroll
      for (int v = 0; v < 8; ++v) a[v] = al[mt * 256 + v];
      o[mt] = wmma_f16(__builtin_bit_cast(v16h, a), bh, o[mt]);
    }
  }

#pragma unroll
  for (int mt = 0; mt < 8; ++mt) {
#pragma unroll
    for (int r = 0; r < 8; ++r) {
      int c = mt * 16 + r + ((lane & 16) ? 8 : 0);
      int l = ltile * 16 + (lane & 15);
      vraw[((size_t)(nb * 128 + c) << 12) + l] = o[mt][r];
    }
  }
}

// ---------------------------------------------------------------------------
// 3d. training-mode BN stats -> relu(x*scale+shift) affine. [n][C][L] input.
// ---------------------------------------------------------------------------
__global__ void stats_kernel(const float* __restrict__ x, const float* __restrict__ g,
                             const float* __restrict__ b, float* __restrict__ scale,
                             float* __restrict__ shift, int C) {
  int c = blockIdx.x;
  float s = 0.f, s2 = 0.f;
  for (int i = threadIdx.x; i < 4 * L; i += 256) {
    int n = i >> 12, l = i & (L - 1);
    float v = x[((size_t)(n * C + c) << 12) + l];
    s += v;
    s2 += v * v;
  }
  __shared__ float sh[256], sh2[256];
  sh[threadIdx.x] = s;
  sh2[threadIdx.x] = s2;
  __syncthreads();
  for (int st = 128; st > 0; st >>= 1) {
    if ((int)threadIdx.x < st) {
      sh[threadIdx.x] += sh[threadIdx.x + st];
      sh2[threadIdx.x] += sh2[threadIdx.x + st];
    }
    __syncthreads();
  }
  if (threadIdx.x == 0) {
    float m = sh[0] * (1.f / 16384.f);
    float var = sh2[0] * (1.f / 16384.f) - m * m;
    float rs = rsqrtf(var + 1e-5f);
    scale[c] = g[c] * rs;
    shift[c] = b[c] - m * g[c] * rs;
  }
}

// ---------------------------------------------------------------------------
// 3e. Vop: BN+ReLU value -> B-operand of PV wmma.
//     Layout [n][kc=128][nch=8][lane=32][v=8] dwords (8KB contiguous per kc).
// ---------------------------------------------------------------------------
__global__ void vop_kernel(const float* __restrict__ vraw, const float* __restrict__ scale,
                           const float* __restrict__ shift, int* __restrict__ Vop) {
  int idx = blockIdx.x * 256 + threadIdx.x;
  int c = idx & 127;
  int mp = (idx >> 7) & 2047;
  int n = idx >> 18;
  int m0 = mp * 2;
  float sc = scale[c], sf = shift[c];
  const float* vr = vraw + ((size_t)(n * 128 + c) << 12);
  float a0 = fmaxf(vr[m0] * sc + sf, 0.f);
  float a1 = fmaxf(vr[m0 + 1] * sc + sf, 0.f);
  int kc = m0 >> 5, kk = m0 & 31;
  int lane = (c & 15) + ((kk & 8) ? 16 : 0);
  int v = ((kk & 7) >> 1) + ((kk >= 16) ? 4 : 0);
  int nch = c >> 4;
  Vop[(((size_t)(n * 128 + kc) * 8 + nch) * 32 + lane) * 8 + v] = pack2h(a0, a1);
}

// ---------------------------------------------------------------------------
// 4. Flash attention core (unchanged from round 2; compiles to wmma + async).
// ---------------------------------------------------------------------------
__global__ __launch_bounds__(256) void attn_kernel(const int* __restrict__ Qop,
                                                   const int* __restrict__ Kop,
                                                   const int* __restrict__ Vop,
                                                   float* __restrict__ ctx) {
  __shared__ int lds[2 * 2560];  // [2 bufs][K:512 | V:2048] dwords = 20KB

  const int lane = threadIdx.x & 31;
  const int wave = threadIdx.x >> 5;
  const int nb = blockIdx.x >> 4;
  const int qg0 = (blockIdx.x & 15) * 16 + wave * 2;

  const unsigned ldsbase = lds_off(&lds[0]);

  auto issueK = [&](int kc, int b) {
    if (wave < 4) {
      unsigned long long g = (unsigned long long)(Kop + (size_t)(nb * 256 + kc * 2) * 256)
                             + (unsigned)(wave * 512 + lane * 16);
      async_ld_b128(ldsbase + (unsigned)(b * 10240 + wave * 512 + lane * 16), g);
    }
  };
  auto issueV = [&](int kc, int b) {
    unsigned long long g = (unsigned long long)(Vop + (size_t)(nb * 128 + kc) * 2048)
                           + (unsigned)(wave * 1024 + lane * 16);
    unsigned l0 = ldsbase + (unsigned)(b * 10240 + 2048 + wave * 1024 + lane * 16);
    async_ld_b128(l0, g);
    async_ld_b128(l0 + 512, g + 512);
  };

  v16h qh[2];
#pragma unroll
  for (int t = 0; t < 2; ++t) {
    v8i qi;
    const int* qp = Qop + ((size_t)(nb * 256 + qg0 + t) * 32 + lane) * 8;
#pragma unroll
    for (int v = 0; v < 8; ++v) qi[v] = qp[v];
    qh[t] = __builtin_bit_cast(v16h, qi);
  }

  float mi[2] = {-1e30f, -1e30f};
  issueK(0, 0);
#pragma unroll 1
  for (int kc = 0; kc < 128; ++kc) {
    int b = kc & 1;
    wait_async0();
    __syncthreads();
    if (kc < 127) issueK(kc + 1, 1 - b);
    const int* kl = &lds[b * 2560 + lane * 8];
    v8i k0, k1;
#pragma unroll
    for (int v = 0; v < 8; ++v) { k0[v] = kl[v]; k1[v] = kl[256 + v]; }
#pragma unroll
    for (int t = 0; t < 2; ++t) {
      v8f z = {};
      v8f s0 = wmma_f16(__builtin_bit_cast(v16h, k0), qh[t], z);
      v8f s1 = wmma_f16(__builtin_bit_cast(v16h, k1), qh[t], z);
      float m = s0[0];
#pragma unroll
      for (int r = 1; r < 8; ++r) m = fmaxf(m, s0[r]);
#pragma unroll
      for (int r = 0; r < 8; ++r) m = fmaxf(m, s1[r]);
      m = fmaxf(m, __shfl_xor(m, 16, 32));
      mi[t] = fmaxf(mi[t], m);
    }
  }

  float li[2] = {0.f, 0.f};
  v8f o[16];
#pragma unroll
  for (int c = 0; c < 16; ++c) o[c] = (v8f){};

  issueK(0, 0);
  issueV(0, 0);
#pragma unroll 1
  for (int kc = 0; kc < 128; ++kc) {
    int b = kc & 1;
    wait_async0();
    __syncthreads();
    if (kc < 127) { issueK(kc + 1, 1 - b); issueV(kc + 1, 1 - b); }

    const int* kl = &lds[b * 2560 + lane * 8];
    v8i k0, k1;
#pragma unroll
    for (int v = 0; v < 8; ++v) { k0[v] = kl[v]; k1[v] = kl[256 + v]; }

    v16h pa[2];
#pragma unroll
    for (int t = 0; t < 2; ++t) {
      v8f z = {};
      v8f s0 = wmma_f16(__builtin_bit_cast(v16h, k0), qh[t], z);
      v8f s1 = wmma_f16(__builtin_bit_cast(v16h, k1), qh[t], z);
      float p0[8], p1[8], tsum = 0.f;
#pragma unroll
      for (int r = 0; r < 8; ++r) { p0[r] = __expf(s0[r] - mi[t]); tsum += p0[r]; }
#pragma unroll
      for (int r = 0; r < 8; ++r) { p1[r] = __expf(s1[r] - mi[t]); tsum += p1[r]; }
      li[t] += tsum + __shfl_xor(tsum, 16, 32);
      v8i pai;
#pragma unroll
      for (int v = 0; v < 4; ++v) pai[v] = pack2h(p0[2 * v], p0[2 * v + 1]);
#pragma unroll
      for (int v = 0; v < 4; ++v) pai[4 + v] = pack2h(p1[2 * v], p1[2 * v + 1]);
      pa[t] = __builtin_bit_cast(v16h, pai);
    }

    const int* vl = &lds[b * 2560 + 512 + lane * 8];
#pragma unroll
    for (int c = 0; c < 8; ++c) {
      v8i vb;
#pragma unroll
      for (int v = 0; v < 8; ++v) vb[v] = vl[c * 256 + v];
      v16h vh = __builtin_bit_cast(v16h, vb);
      o[c] = wmma_f16(pa[0], vh, o[c]);
      o[8 + c] = wmma_f16(pa[1], vh, o[8 + c]);
    }
  }

#pragma unroll
  for (int t = 0; t < 2; ++t) {
#pragma unroll
    for (int r = 0; r < 8; ++r) {
      int src = r + ((lane & 16) ? 8 : 0);
      float ls = __shfl(li[t], src, 32);
      float rc = 1.0f / ls;
      int qrow = (qg0 + t) * 16 + r + ((lane & 16) ? 8 : 0);
      float* op = ctx + ((size_t)(nb * L + qrow) * 128) + (lane & 15);
#pragma unroll
      for (int c = 0; c < 8; ++c) op[c * 16] = o[t * 8 + c][r] * rc;
    }
  }
}

// ---------------------------------------------------------------------------
// 5. proj[n][10][L] = Wp[10,128] @ ctx
// ---------------------------------------------------------------------------
__global__ void proj_kernel(const float* __restrict__ ctx, const float* __restrict__ Wp,
                            float* __restrict__ proj) {
  int idx = blockIdx.x * 256 + threadIdx.x;
  if (idx >= 4 * 10 * L) return;
  int l = idx & (L - 1);
  int t = idx >> 12;
  int ochan = t % 10, n = t / 10;
  const float* cp = ctx + ((size_t)(n * L) + l) * 128;
  const float* w = Wp + ochan * 128;
  float s = 0.f;
#pragma unroll 4
  for (int c = 0; c < 128; ++c) s += w[c] * cp[c];
  proj[idx] = s;
}

// ---------------------------------------------------------------------------
// 6. Fused BN+ReLU + bilinear resize 64->129 (align_corners, scale 63/128).
// ---------------------------------------------------------------------------
__global__ void resize_kernel(const float* __restrict__ proj, const float* __restrict__ scale,
                              const float* __restrict__ shift, float* __restrict__ out) {
  int idx = blockIdx.x * 256 + threadIdx.x;
  if (idx >= 4 * 10 * 129 * 129) return;
  int xy = idx % 16641;
  int t = idx / 16641;
  int o = t % 10, n = t / 10;
  int X = xy % 129, Y = xy / 129;
  float fy = Y * (63.f / 128.f);
  float fx = X * (63.f / 128.f);
  int y0 = (int)fy, x0 = (int)fx;
  int y1 = min(y0 + 1, 63), x1 = min(x0 + 1, 63);
  float wy = fy - (float)y0, wx = fx - (float)x0;
  const float* pp = proj + ((size_t)(n * 10 + o) << 12);
  float sc = scale[o], sf = shift[o];
  float f00 = fmaxf(pp[y0 * 64 + x0] * sc + sf, 0.f);
  float f01 = fmaxf(pp[y0 * 64 + x1] * sc + sf, 0.f);
  float f10 = fmaxf(pp[y1 * 64 + x0] * sc + sf, 0.f);
  float f11 = fmaxf(pp[y1 * 64 + x1] * sc + sf, 0.f);
  out[idx] = f00 * (1 - wy) * (1 - wx) + f01 * (1 - wy) * wx +
             f10 * wy * (1 - wx) + f11 * wy * wx;
}

// ---------------------------------------------------------------------------
extern "C" void kernel_launch(void* const* d_in, const int* in_sizes, int n_in,
                              void* d_out, int out_size, void* d_ws, size_t ws_size,
                              hipStream_t stream) {
  const float* p_fea = (const float*)d_in[0];
  const float* hu    = (const float*)d_in[1];
  const float* Wq    = (const float*)d_in[2];
  const float* Wk    = (const float*)d_in[3];
  const float* Wv    = (const float*)d_in[4];
  const float* g_v   = (const float*)d_in[5];
  const float* b_v   = (const float*)d_in[6];
  const float* Wp    = (const float*)d_in[7];
  const float* g_p   = (const float*)d_in[8];
  const float* b_p   = (const float*)d_in[9];
  float* out = (float*)d_out;

  char* ws = (char*)d_ws;
  size_t off = 0;
  auto alloc = [&](size_t dwords) {
    void* r = ws + off;
    off += ((dwords * 4 + 255) & ~(size_t)255);
    return r;
  };
  float* p      = (float*)alloc((size_t)4 * 256 * L);
  float* u      = (float*)alloc((size_t)4 * 10 * L);
  int*   Qop    = (int*)  alloc((size_t)4 * 256 * 32 * 8);
  int*   Kop    = (int*)  alloc((size_t)4 * 256 * 32 * 8);
  int*   Vop    = (int*)  alloc((size_t)4 * 128 * 8 * 32 * 8);
  int*   Wop    = (int*)  alloc((size_t)8 * 8 * 32 * 8);
  int*   Pop    = (int*)  alloc((size_t)4 * 8 * 256 * 32 * 8);
  float* vraw   = (float*)alloc((size_t)4 * 128 * L);
  float* vscale = (float*)alloc(128);
  float* vshift = (float*)alloc(128);
  float* ctx    = (float*)alloc((size_t)4 * L * 128);
  float* proj   = (float*)alloc((size_t)4 * 10 * L);
  float* pscale = (float*)alloc(16);
  float* pshift = (float*)alloc(16);
  (void)ws_size; (void)in_sizes; (void)n_in; (void)out_size;

  pool_kernel<<<(4 * 256 * L + 255) / 256, 256, 0, stream>>>(p_fea, p, 4 * 256);
  pool_kernel<<<(4 * 10 * L + 255) / 256, 256, 0, stream>>>(hu, u, 4 * 10);

  qkprep_kernel<<<(4 * L + 255) / 256, 256, 0, stream>>>(u, p, Wq, Wk, Qop, Kop);

  // value path: pack -> WMMA GEMM (TDM-staged A) -> BN stats -> operand pack
  wpack_kernel<<<64, 256, 0, stream>>>(Wv, Wop);
  ppack_kernel<<<8192, 256, 0, stream>>>(p, Pop);
  vgemm_kernel<<<128, 256, 0, stream>>>(Wop, Pop, vraw);
  stats_kernel<<<128, 256, 0, stream>>>(vraw, g_v, b_v, vscale, vshift, 128);
  vop_kernel<<<(4 * 2048 * 128) / 256, 256, 0, stream>>>(vraw, vscale, vshift, Vop);

  attn_kernel<<<64, 256, 0, stream>>>(Qop, Kop, Vop, ctx);

  proj_kernel<<<(4 * 10 * L + 255) / 256, 256, 0, stream>>>(ctx, Wp, proj);
  stats_kernel<<<10, 256, 0, stream>>>(proj, g_p, b_p, pscale, pshift, 10);

  resize_kernel<<<(4 * 10 * 129 * 129 + 255) / 256, 256, 0, stream>>>(proj, pscale, pshift, out);
}
